// GATNetwork_69346541961378
// MI455X (gfx1250) — compile-verified, matrix operand order checked by reading, not run on hardware
//
#include <hip/hip_runtime.h>
#include <hip/hip_bf16.h>

#define N_NODES 50000
#define N_EDGES 800000
#define F_IN    64
#define HD      128
#define GB      8
#define NA      16

typedef __attribute__((ext_vector_type(16))) _Float16 v16h;
typedef __attribute__((ext_vector_type(4)))  _Float16 v4h;
typedef __attribute__((ext_vector_type(8)))  float    v8f;
typedef __attribute__((ext_vector_type(4)))  int      v4i;

#if __has_builtin(__builtin_amdgcn_global_load_async_to_lds_b128) && \
    __has_builtin(__builtin_amdgcn_s_wait_asynccnt)
#define HAVE_ASYNC_LDS 1
#else
#define HAVE_ASYNC_LDS 0
#endif

static inline int cdiv(int a, int b){ return (a + b - 1) / b; }

// ---------------------------------------------------------------- utilities

__global__ void fill_kernel(float* __restrict__ p, int n, float v){
    int i = blockIdx.x * blockDim.x + threadIdx.x;
    if (i < n) p[i] = v;
}

__device__ __forceinline__ float atomicMaxF(float* addr, float val){
    // IEEE trick: positive floats order as ints, negative as reversed uints.
    if (val >= 0.0f)
        return __int_as_float(atomicMax((int*)addr, __float_as_int(val)));
    return __uint_as_float(atomicMin((unsigned int*)addr, __float_as_uint(val)));
}

// ---------------------------------------------------------------- W prep
// Swizzle W (K x 128, f32 row-major) into WMMA B-fragment order as f16,
// ONCE per GEMM, so every block's LDS staging is a contiguous copy.
//   Wf[((kt*8 + t)*32 + lane)*16 + j] = f16(W[k][n])
//     n = t*16 + (lane&15), kh = lane>>4,
//     k = kt*32 + ((j>>3)<<4) + kh*8 + (j&7)          (ISA 16-bit B order)
__global__ void prep_w_kernel(const float* __restrict__ W, _Float16* __restrict__ Wf,
                              int K){
    int i = blockIdx.x * blockDim.x + threadIdx.x;
    if (i >= K * HD) return;
    int j  = i & 15;
    int ln = (i >> 4) & 31;
    int t  = (i >> 9) & 7;
    int kt = i >> 12;
    int n  = t * 16 + (ln & 15);
    int kh = ln >> 4;
    int k  = kt * 32 + ((j >> 3) << 4) + kh * 8 + (j & 7);
    Wf[i] = (_Float16)W[k * HD + n];
}

// ---------------------------------------------------------------- WMMA GEMM
// Y[r, j] = sum_k X[r,k] * W[k,j] + bias[j]     (H = 128, K in {64,128})
// Block: 128 threads (4 waves); 64-row x 128-col output tile per block.
// Wave w handles rows [w*16, w*16+16) -> 8 accumulators of 16x16 f32.
// Wf is pre-swizzled, so both WMMA operands are single per-lane v16h reads.
__global__ __launch_bounds__(128)
void gemm_bias_wmma(const float* __restrict__ X, const _Float16* __restrict__ Wf,
                    const float* __restrict__ bias, float* __restrict__ Y,
                    int nrows, int K)
{
    __shared__ _Float16 sW[HD * HD];   // 32 KB max (of 320 KB WGP LDS)
    __shared__ _Float16 sA[64 * 32];   // 4 KB

    const int tid  = threadIdx.x;
    const int lane = tid & 31;
    const int wave = tid >> 5;
    const int rowBase = blockIdx.x * 64;

    // ---- stage Wf -> LDS: contiguous block copy (fragment order already) ----
#if HAVE_ASYNC_LDS
    {
        const int n16 = (K * HD * 2) / 16;             // 16B chunks
        auto gsrc = (__attribute__((address_space(1))) v4i*)(unsigned long long)
                    (const char*)Wf;
        auto ldst = (__attribute__((address_space(3))) v4i*)sW;
        for (int i = tid; i < n16; i += 128)
            __builtin_amdgcn_global_load_async_to_lds_b128(gsrc + i, ldst + i, 0, 0);
        __builtin_amdgcn_s_wait_asynccnt(0);
    }
#else
    {
        const uint4* src = (const uint4*)Wf;           // 8 f16 per uint4
        uint4* dst = (uint4*)sW;
        for (int i = tid; i < K * HD / 8; i += 128) dst[i] = src[i];
    }
#endif

    v8f acc[8] = {};

    const int m16 = lane & 15;   // row (A) / col (B) within 16
    const int kh  = lane >> 4;   // K-half selector per ISA 16-bit layout

    const int nkt = K >> 5;
    for (int kt = 0; kt < nkt; ++kt){
        // ---- stage A tile (64 rows x 32 K): coalesced float4 reads,
        //      packed 4xf16 ds_store_b64, swizzled into A-fragment order ----
        for (int g = tid; g < (64 * 32) / 4; g += 128){
            int i   = g * 4;            // element index, 4-aligned
            int r   = i >> 5;
            int kr  = i & 31;           // source k within chunk (contiguous)
            int j   = (kr & 7) | ((kr >> 4) << 3);
            int akh = (kr >> 3) & 1;
            int row = rowBase + r;
            float4 v = make_float4(0.f, 0.f, 0.f, 0.f);
            if (row < nrows)
                v = *(const float4*)(X + (size_t)row * K + kt * 32 + kr);
            if (kt + 1 < nkt && row < nrows)
                __builtin_prefetch(X + (size_t)row * K + (kt + 1) * 32 + kr, 0, 1);
            *(v4h*)(sA + (r * 32 + akh * 16 + j)) =
                (v4h){(_Float16)v.x, (_Float16)v.y, (_Float16)v.z, (_Float16)v.w};
        }
        __syncthreads();

        // A fragment: contiguous per-lane 32B chunk
        v16h a = *(const v16h*)(sA + ((wave * 16 + m16) * 32 + kh * 16));

        #pragma unroll
        for (int t = 0; t < 8; ++t){
            // B fragment: contiguous per-lane 32B chunk (pre-swizzled)
            v16h b = *(const v16h*)(sW + (((kt * 8 + t) * 32 + lane) * 16));
            acc[t] = __builtin_amdgcn_wmma_f32_16x16x32_f16(
                false, a, false, b, (short)0, acc[t], false, false);
        }
        __syncthreads();
    }

    // C/D layout: VGPR i, lanes 0-15 -> M=i, lanes 16-31 -> M=i+8; N = m16
    #pragma unroll
    for (int t = 0; t < 8; ++t){
        int col = t * 16 + m16;
        float bb = bias[col];
        #pragma unroll
        for (int i = 0; i < 8; ++i){
            int row = rowBase + wave * 16 + kh * 8 + i;
            if (row < nrows) Y[(size_t)row * HD + col] = acc[t][i] + bb;
        }
    }
}

// ---------------------------------------------------------------- edge phase

#define LRELU2(v) ((v) > 0.0f ? (v) : 0.2f * (v))

// wave per edge: e = leaky(xl[src]+xr[dst]+edge_attr@We); logits = e@att
__global__ void edge_logits_kernel(const int* __restrict__ ei,
                                   const float* __restrict__ eattr,
                                   const float* __restrict__ xl,
                                   const float* __restrict__ xr,
                                   const float* __restrict__ We,
                                   const float* __restrict__ att,
                                   float* __restrict__ logits,
                                   float* __restrict__ mmax)
{
    int e    = (blockIdx.x * blockDim.x + threadIdx.x) >> 5;
    int lane = threadIdx.x & 31;
    if (e >= N_EDGES) return;
    int s = ei[e], d = ei[N_EDGES + e];
    float ea0 = eattr[2 * e], ea1 = eattr[2 * e + 1];

    const float4* xl4 = (const float4*)(xl + (size_t)s * HD);
    const float4* xr4 = (const float4*)(xr + (size_t)d * HD);
    const float4* w04 = (const float4*)We;
    const float4* w14 = (const float4*)(We + HD);
    const float4* at4 = (const float4*)att;

    float4 va = xl4[lane], vb = xr4[lane];
    float4 w0 = w04[lane], w1 = w14[lane], at = at4[lane];

    float part;
    part  = LRELU2(va.x + vb.x + ea0 * w0.x + ea1 * w1.x) * at.x;
    part += LRELU2(va.y + vb.y + ea0 * w0.y + ea1 * w1.y) * at.y;
    part += LRELU2(va.z + vb.z + ea0 * w0.z + ea1 * w1.z) * at.z;
    part += LRELU2(va.w + vb.w + ea0 * w0.w + ea1 * w1.w) * at.w;

    #pragma unroll
    for (int off = 16; off > 0; off >>= 1) part += __shfl_xor(part, off);

    if (lane == 0){
        logits[e] = part;
        atomicMaxF(&mmax[d], part);
    }
}

// thread per edge: p = exp(logit - max[dst]); denom[dst] += p
__global__ void edge_softmax_num(const int* __restrict__ ei,
                                 float* __restrict__ logits,
                                 const float* __restrict__ mmax,
                                 float* __restrict__ denom)
{
    int e = blockIdx.x * blockDim.x + threadIdx.x;
    if (e >= N_EDGES) return;
    int d = ei[N_EDGES + e];
    float p = __expf(logits[e] - mmax[d]);
    logits[e] = p;
    atomicAdd(&denom[d], p);
}

// wave per edge: out[dst] += alpha * xl[src]
__global__ void edge_scatter_kernel(const int* __restrict__ ei,
                                    const float* __restrict__ logits,
                                    const float* __restrict__ denom,
                                    const float* __restrict__ xl,
                                    float* __restrict__ out)
{
    int e    = (blockIdx.x * blockDim.x + threadIdx.x) >> 5;
    int lane = threadIdx.x & 31;
    if (e >= N_EDGES) return;
    int s = ei[e], d = ei[N_EDGES + e];
    float alpha = logits[e] / (denom[d] + 1e-16f);
    float4 v = ((const float4*)(xl + (size_t)s * HD))[lane];
    float* o = out + (size_t)d * HD + lane * 4;
    atomicAdd(o + 0, alpha * v.x);
    atomicAdd(o + 1, alpha * v.y);
    atomicAdd(o + 2, alpha * v.z);
    atomicAdd(o + 3, alpha * v.w);
}

__global__ void add_bias_kernel(float* __restrict__ h, const float* __restrict__ bias){
    int i = blockIdx.x * blockDim.x + threadIdx.x;
    if (i < N_NODES * HD) h[i] += bias[i & (HD - 1)];
}

// ---------------------------------------------------------------- GraphNorm

__global__ void gn_count_kernel(const int* __restrict__ batch, float* __restrict__ gcnt){
    int n = blockIdx.x * blockDim.x + threadIdx.x;
    if (n < N_NODES) atomicAdd(&gcnt[batch[n]], 1.0f);
}

__global__ void gn_sum_kernel(const float* __restrict__ h, const int* __restrict__ batch,
                              float* __restrict__ gsum){
    int i = blockIdx.x * blockDim.x + threadIdx.x;
    if (i < N_NODES * HD){
        int n = i >> 7, c = i & (HD - 1);
        atomicAdd(&gsum[batch[n] * HD + c], h[i]);
    }
}

__global__ void gn_div_kernel(float* __restrict__ gs, const float* __restrict__ gcnt){
    int i = blockIdx.x * blockDim.x + threadIdx.x;
    if (i < GB * HD) gs[i] /= gcnt[i >> 7];
}

__global__ void gn_sq_kernel(const float* __restrict__ h, const int* __restrict__ batch,
                             const float* __restrict__ mean, const float* __restrict__ ga,
                             float* __restrict__ gsq){
    int i = blockIdx.x * blockDim.x + threadIdx.x;
    if (i < N_NODES * HD){
        int n = i >> 7, c = i & (HD - 1);
        int g = batch[n];
        float xc = h[i] - ga[c] * mean[g * HD + c];
        atomicAdd(&gsq[g * HD + c], xc * xc);
    }
}

__global__ void gn_norm_kernel(const float* __restrict__ h, const int* __restrict__ batch,
                               const float* __restrict__ mean, const float* __restrict__ var,
                               const float* __restrict__ gw, const float* __restrict__ gb,
                               const float* __restrict__ ga, float* __restrict__ out){
    int i = blockIdx.x * blockDim.x + threadIdx.x;
    if (i < N_NODES * HD){
        int n = i >> 7, c = i & (HD - 1);
        int g = batch[n];
        float xc = h[i] - ga[c] * mean[g * HD + c];
        float y  = gw[c] * xc * rsqrtf(var[g * HD + c] + 1e-5f) + gb[c];
        out[i] = y > 0.0f ? y : 0.01f * y;
    }
}

// ---------------------------------------------------------------- pool + head

__global__ void pool_kernel(const float* __restrict__ h, const int* __restrict__ batch,
                            float* __restrict__ pooled){
    int i = blockIdx.x * blockDim.x + threadIdx.x;
    if (i < N_NODES * HD){
        int n = i >> 7, c = i & (HD - 1);
        atomicAdd(&pooled[batch[n] * HD + c], h[i]);
    }
}

__global__ __launch_bounds__(128)
void head_kernel(const float* __restrict__ pooled,
                 const float* __restrict__ W1, const float* __restrict__ b1,
                 const float* __restrict__ W2, const float* __restrict__ b2,
                 float* __restrict__ out)
{
    __shared__ float hid[GB * HD];
    int tid = threadIdx.x;               // 128 threads: one hidden column each
    for (int g = 0; g < GB; ++g){
        float acc = b1[tid];
        for (int k = 0; k < HD; ++k) acc += pooled[g * HD + k] * W1[k * HD + tid];
        hid[g * HD + tid] = acc > 0.0f ? acc : 0.01f * acc;
    }
    __syncthreads();
    int g = tid >> 4, a = tid & (NA - 1); // 8 graphs x 16 actions = 128 outputs
    float acc = b2[a];
    for (int j = 0; j < HD; ++j) acc += hid[g * HD + j] * W2[j * NA + a];
    out[g * NA + a] = acc;
}

// ---------------------------------------------------------------- launch

extern "C" void kernel_launch(void* const* d_in, const int* in_sizes, int n_in,
                              void* d_out, int out_size, void* d_ws, size_t ws_size,
                              hipStream_t stream)
{
    (void)in_sizes; (void)n_in; (void)out_size; (void)ws_size;

    const float* x     = (const float*)d_in[0];
    const int*   ei    = (const int*)  d_in[1];
    const float* eattr = (const float*)d_in[2];
    const int*   batch = (const int*)  d_in[3];

    float* ws = (float*)d_ws;
    float* bufXL  = ws; ws += (size_t)N_NODES * HD;
    float* bufXR  = ws; ws += (size_t)N_NODES * HD;
    float* bufOut = ws; ws += (size_t)N_NODES * HD;
    float* bufH   = ws; ws += (size_t)N_NODES * HD;
    float* mmax   = ws; ws += N_NODES;
    float* denom  = ws; ws += N_NODES;
    float* logits = ws; ws += N_EDGES;
    float* gsum   = ws; ws += GB * HD;
    float* gsq    = ws; ws += GB * HD;
    float* gcnt   = ws; ws += GB;
    float* pooled = ws; ws += GB * HD;
    _Float16* wfrag = (_Float16*)ws; ws += (HD * HD) / 2;  // 16K f16, 16B aligned

    const int nbNH   = cdiv(N_NODES * HD, 256);
    const int nbN    = cdiv(N_NODES, 256);
    const int nbEw   = cdiv(N_EDGES * 32, 256);   // wave-per-edge kernels
    const int nbE    = cdiv(N_EDGES, 256);
    const int nbGH   = cdiv(GB * HD, 256);
    const int nbGemm = cdiv(N_NODES, 64);

    for (int l = 0; l < 3; ++l){
        const float* Wl   = (const float*)d_in[4 + l * 7 + 0];
        const float* bl   = (const float*)d_in[4 + l * 7 + 1];
        const float* Wr   = (const float*)d_in[4 + l * 7 + 2];
        const float* br   = (const float*)d_in[4 + l * 7 + 3];
        const float* We   = (const float*)d_in[4 + l * 7 + 4];
        const float* att  = (const float*)d_in[4 + l * 7 + 5];
        const float* bias = (const float*)d_in[4 + l * 7 + 6];

        const float* Xin = (l == 0) ? x : bufH;
        int K = (l == 0) ? F_IN : HD;
        int nbPW = cdiv(K * HD, 256);

        prep_w_kernel<<<nbPW, 256, 0, stream>>>(Wl, wfrag, K);
        gemm_bias_wmma<<<nbGemm, 128, 0, stream>>>(Xin, wfrag, bl, bufXL, N_NODES, K);
        prep_w_kernel<<<nbPW, 256, 0, stream>>>(Wr, wfrag, K);
        gemm_bias_wmma<<<nbGemm, 128, 0, stream>>>(Xin, wfrag, br, bufXR, N_NODES, K);

        fill_kernel<<<nbN,  256, 0, stream>>>(mmax,  N_NODES, -3.0e38f);
        fill_kernel<<<nbN,  256, 0, stream>>>(denom, N_NODES, 0.0f);
        fill_kernel<<<nbNH, 256, 0, stream>>>(bufOut, N_NODES * HD, 0.0f);

        edge_logits_kernel<<<nbEw, 256, 0, stream>>>(ei, eattr, bufXL, bufXR, We, att,
                                                     logits, mmax);
        edge_softmax_num<<<nbE, 256, 0, stream>>>(ei, logits, mmax, denom);
        edge_scatter_kernel<<<nbEw, 256, 0, stream>>>(ei, logits, denom, bufXL, bufOut);
        add_bias_kernel<<<nbNH, 256, 0, stream>>>(bufOut, bias);

        if (l < 2){
            const float* gw = (const float*)d_in[25 + l * 3 + 0];
            const float* gb = (const float*)d_in[25 + l * 3 + 1];
            const float* ga = (const float*)d_in[25 + l * 3 + 2];

            fill_kernel<<<nbGH, 256, 0, stream>>>(gsum, GB * HD, 0.0f);
            fill_kernel<<<nbGH, 256, 0, stream>>>(gsq,  GB * HD, 0.0f);
            fill_kernel<<<1,    256, 0, stream>>>(gcnt, GB, 0.0f);
            gn_count_kernel<<<nbN,  256, 0, stream>>>(batch, gcnt);
            gn_sum_kernel  <<<nbNH, 256, 0, stream>>>(bufOut, batch, gsum);
            gn_div_kernel  <<<nbGH, 256, 0, stream>>>(gsum, gcnt);
            gn_sq_kernel   <<<nbNH, 256, 0, stream>>>(bufOut, batch, gsum, ga, gsq);
            gn_div_kernel  <<<nbGH, 256, 0, stream>>>(gsq, gcnt);
            gn_norm_kernel <<<nbNH, 256, 0, stream>>>(bufOut, batch, gsum, gsq,
                                                      gw, gb, ga, bufH);
        }
    }

    fill_kernel<<<nbGH, 256, 0, stream>>>(pooled, GB * HD, 0.0f);
    pool_kernel<<<nbNH, 256, 0, stream>>>(bufOut, batch, pooled);
    head_kernel<<<1, 128, 0, stream>>>(pooled,
                                       (const float*)d_in[31], (const float*)d_in[32],
                                       (const float*)d_in[33], (const float*)d_in[34],
                                       (float*)d_out);
}
